// EnhancedEdgeRankingGNN_41875931136406
// MI455X (gfx1250) — matrix operand order repaired and checked by reading, not compiled, not run on hardware
//
#include <hip/hip_runtime.h>
#include <hip/hip_bf16.h>

#define N_NODES 50000
#define N_EDGES 400000
#define NGRAPH  64
#define H       128

typedef __attribute__((ext_vector_type(16))) _Float16 v16h;
typedef __attribute__((ext_vector_type(8)))  _Float16 v8h;
typedef __attribute__((ext_vector_type(8)))  float    v8f;

// ---------------- WMMA fragment helpers (wave32, 16x16x32 f16) ----------------
// A: 16x32 f16 tile from row-major matrix (row stride lda). Per ISA layout:
// lanes 0-15 hold rows M=0..15, K=[0..7]+[16..23]; lanes 16-31 K=[8..15]+[24..31].
static __device__ __forceinline__ v16h load_a_frag(const _Float16* A, int lda, int lane, int kc)
{
    int m  = lane & 15;
    int kb = (lane >> 4) * 8;
    const _Float16* p = A + (size_t)m * lda + kc * 32 + kb;
    v8h lo = *(const v8h*)(p);
    v8h hi = *(const v8h*)(p + 16);
    v16h a;
#pragma unroll
    for (int j = 0; j < 8; ++j) { a[j] = lo[j]; a[j + 8] = hi[j]; }
    return a;
}

// B fragments are pre-formatted: [ntile][kchunk][lane][16] contiguous halves.
static __device__ __forceinline__ v16h load_b_frag(const _Float16* frag, int lane, int nt, int kc, int kchunks)
{
    return *(const v16h*)(frag + ((((size_t)nt * kchunks + kc) * 32) + lane) * 16);
}

// C/D layout: VGPR r -> M = r + (lane>=16 ? 8 : 0), N = (lane&15) + 16*ntile.
template <int ACT> // 0=none, 1=relu, 2=tanh
static __device__ __forceinline__ void gemm_tile_f16(
    const _Float16* A, int Kd, const _Float16* Bfrag, const float* bias,
    _Float16* Out, int Nd, int lane)
{
    const int kchunks = Kd >> 5;
    const int ntiles  = Nd >> 4;
    for (int nt = 0; nt < ntiles; ++nt) {
        v8f c = {};
        for (int kc = 0; kc < kchunks; ++kc) {
            v16h a = load_a_frag(A, Kd, lane, kc);
            v16h b = load_b_frag(Bfrag, lane, nt, kc, kchunks);
            c = __builtin_amdgcn_wmma_f32_16x16x32_f16(false, a, false, b, (short)0, c, false, false);
        }
        int n  = (lane & 15) + nt * 16;
        int mb = (lane >> 4) * 8;
        float bv = bias ? bias[n] : 0.0f;
#pragma unroll
        for (int r = 0; r < 8; ++r) {
            float v = c[r] + bv;
            if (ACT == 1) v = fmaxf(v, 0.0f);
            if (ACT == 2) v = tanhf(v);
            Out[(size_t)(mb + r) * Nd + n] = (_Float16)v;
        }
    }
}

template <int ACT>
static __device__ __forceinline__ void gemm_tile_f32(
    const _Float16* A, int Kd, const _Float16* Bfrag, const float* bias,
    float* Out, int Nd, int lane)
{
    const int kchunks = Kd >> 5;
    const int ntiles  = Nd >> 4;
    for (int nt = 0; nt < ntiles; ++nt) {
        v8f c = {};
        for (int kc = 0; kc < kchunks; ++kc) {
            v16h a = load_a_frag(A, Kd, lane, kc);
            v16h b = load_b_frag(Bfrag, lane, nt, kc, kchunks);
            c = __builtin_amdgcn_wmma_f32_16x16x32_f16(false, a, false, b, (short)0, c, false, false);
        }
        int n  = (lane & 15) + nt * 16;
        int mb = (lane >> 4) * 8;
        float bv = bias ? bias[n] : 0.0f;
#pragma unroll
        for (int r = 0; r < 8; ++r) {
            float v = c[r] + bv;
            if (ACT == 1) v = fmaxf(v, 0.0f);
            if (ACT == 2) v = tanhf(v);
            Out[(size_t)(mb + r) * Nd + n] = v;
        }
    }
}

// LayerNorm over 16 rows of H=128 held in LDS f32; lanes 0..15 each own a row.
static __device__ __forceinline__ void layernorm16(
    const float* S, const float* g, const float* b,
    _Float16* dst, int dst_stride, int maxrow, int lane)
{
    if (lane < maxrow) {
        const float* row = S + lane * H;
        float mu = 0.0f;
        for (int j = 0; j < H; ++j) mu += row[j];
        mu *= (1.0f / H);
        float var = 0.0f;
        for (int j = 0; j < H; ++j) { float d = row[j] - mu; var += d * d; }
        var *= (1.0f / H);
        float inv = rsqrtf(var + 1e-5f);
        for (int j = 0; j < H; ++j) {
            float v = (row[j] - mu) * inv * g[j] + b[j];
            dst[(size_t)lane * dst_stride + j] = (_Float16)v;
        }
    }
}

// ---------------- weight pre-formatting: f32 row-major [K,Nd] -> B fragments f16
__global__ void k_prep_bfrag(const float* W, _Float16* frag, int K, int Nd)
{
    int idx = blockIdx.x * blockDim.x + threadIdx.x;
    int kchunks = (K + 31) >> 5, ntiles = (Nd + 15) >> 4;
    int total = ntiles * kchunks * 32;
    if (idx >= total) return;
    int lane = idx & 31;
    int t = idx >> 5;
    int kc = t % kchunks;
    int nt = t / kchunks;
    int nn = (lane & 15) + nt * 16;
    int kb = kc * 32 + (lane >> 4) * 8;
    v16h v;
#pragma unroll
    for (int j = 0; j < 16; ++j) {
        int k = kb + (j < 8 ? j : 8 + j);
        v[j] = (k < K && nn < Nd) ? (_Float16)W[(size_t)k * Nd + nn] : (_Float16)0.0f;
    }
    *(v16h*)(frag + (size_t)idx * 16) = v;
}

__global__ void k_zero(float* p, long cnt)
{
    long i = (long)blockIdx.x * blockDim.x + threadIdx.x;
    if (i < cnt) p[i] = 0.0f;
}

// ---------------- node encoder: x[N,3] -> h[N,128] f16 (MLP + LN)
__global__ void k_node_enc(
    const float* x,
    const float* w1, const float* b1,
    const _Float16* f2, const float* b2,
    const _Float16* f3, const float* b3,
    const float* g, const float* bb,
    _Float16* out, int n)
{
    __shared__ _Float16 s1[16 * H];
    __shared__ _Float16 s2[16 * H];
    __shared__ float    sf[16 * H];
    const int tile = blockIdx.x, lane = threadIdx.x;
    const int m = lane & 15, half = (lane >> 4) * 64;
    const int r0 = tile * 16;
    int row = r0 + m;
    int rowc = row < n ? row : (n - 1);
    float x0 = x[(size_t)rowc * 3 + 0];
    float x1 = x[(size_t)rowc * 3 + 1];
    float x2 = x[(size_t)rowc * 3 + 2];
    for (int c = 0; c < 64; ++c) {
        int cc = half + c;
        float v = x0 * w1[0 * H + cc] + x1 * w1[1 * H + cc] + x2 * w1[2 * H + cc] + b1[cc];
        s1[m * H + cc] = (_Float16)fmaxf(v, 0.0f);
    }
    __syncthreads();
    gemm_tile_f16<1>(s1, H, f2, b2, s2, H, lane);
    __syncthreads();
    gemm_tile_f32<0>(s2, H, f3, b3, sf, H, lane);
    __syncthreads();
    int maxrow = n - r0; if (maxrow > 16) maxrow = 16;
    layernorm16(sf, g, bb, out + (size_t)r0 * H, H, maxrow, lane);
}

// ---------------- xw = h @ W (N x 128 x 128), f16 A, fragment B, f32 out
__global__ void k_gemm_h(const _Float16* A, const _Float16* Bfrag, float* C)
{
    const int tile = blockIdx.x, lane = threadIdx.x;
    const _Float16* At = A + (size_t)tile * 16 * H;
    float* Ct = C + (size_t)tile * 16 * H;
    for (int nt = 0; nt < 8; ++nt) {
        v8f c = {};
        for (int kc = 0; kc < 4; ++kc) {
            v16h a = load_a_frag(At, H, lane, kc);
            v16h b = load_b_frag(Bfrag, lane, nt, kc, 4);
            c = __builtin_amdgcn_wmma_f32_16x16x32_f16(false, a, false, b, (short)0, c, false, false);
        }
        int n  = (lane & 15) + nt * 16;
        int mb = (lane >> 4) * 8;
#pragma unroll
        for (int r = 0; r < 8; ++r)
            Ct[(size_t)(mb + r) * H + n] = c[r];
    }
}

// ---------------- degree / dinv ----------------
__global__ void k_deg(const int* dst, float* deg, int e)
{
    int i = blockIdx.x * blockDim.x + threadIdx.x;
    if (i < e) atomicAdd(&deg[dst[i]], 1.0f);
}

__global__ void k_dinv(float* deg, int n)
{
    int i = blockIdx.x * blockDim.x + threadIdx.x;
    if (i < n) deg[i] = rsqrtf(deg[i] + 1.0f);
}

// ---------------- scatter: agg[dst] += xw[src] * dinv[src]*dinv[dst] ----------------
__global__ void k_scatter(const int* src, const int* dst, const float* xw,
                          const float* dinv, float* agg, int e)
{
    long idx = (long)blockIdx.x * blockDim.x + threadIdx.x;
    int eid  = (int)(idx >> 5);
    int lane = (int)(idx & 31);
    if (eid >= e) return;
    int s = src[eid], d = dst[eid];
    float w = dinv[s] * dinv[d];
    const float* xr = xw + (size_t)s * H;
    float* ar = agg + (size_t)d * H;
    for (int c = lane; c < H; c += 32)
        atomicAdd(&ar[c], xr[c] * w);
}

// ---------------- h_next = agg + xw * dinv^2 + bias  (optional relu), f16 out
__global__ void k_gcn_fin(const float* agg, const float* xw, const float* dinv,
                          const float* bias, _Float16* out, int n, int relu)
{
    long i = (long)blockIdx.x * blockDim.x + threadIdx.x;
    if (i >= (long)n * H) return;
    int row = (int)(i >> 7), c = (int)(i & 127);
    float dd = dinv[row];
    float v = agg[i] + xw[i] * dd * dd + bias[c];
    if (relu) v = fmaxf(v, 0.0f);
    out[i] = (_Float16)v;
}

// ---------------- global mean pool: sums[batch]+=h, cnt[batch]+=1 ----------------
__global__ void k_pool(const _Float16* h, const int* batch, float* sums, float* cnt, int n)
{
    long idx = (long)blockIdx.x * blockDim.x + threadIdx.x;
    int node = (int)(idx >> 5);
    int lane = (int)(idx & 31);
    if (node >= n) return;
    int g = batch[node];
    if (lane == 0) atomicAdd(&cnt[g], 1.0f);
    const _Float16* hr = h + (size_t)node * H;
    float* sr = sums + (size_t)g * H;
    for (int c = lane; c < H; c += 32)
        atomicAdd(&sr[c], (float)hr[c]);
}

// ---------------- graph MLP: gf = LN(relu(gmean@gp1+b)@gp2+b) -> f16 [G,128]
__global__ void k_graph(const float* sums, const float* cnt,
                        const _Float16* f1, const float* b1,
                        const _Float16* f2, const float* b2,
                        const float* g, const float* bb, _Float16* gf)
{
    __shared__ _Float16 s1[16 * H];
    __shared__ _Float16 s2[16 * H];
    __shared__ float    sf[16 * H];
    const int tile = blockIdx.x, lane = threadIdx.x;
    const int m = lane & 15, half = (lane >> 4) * 64;
    int row = tile * 16 + m;
    float c = cnt[row]; c = c > 1.0f ? c : 1.0f;
    float inv = 1.0f / c;
    for (int j = 0; j < 64; ++j) {
        int cc = half + j;
        s1[m * H + cc] = (_Float16)(sums[(size_t)row * H + cc] * inv);
    }
    __syncthreads();
    gemm_tile_f16<1>(s1, H, f1, b1, s2, H, lane);
    __syncthreads();
    gemm_tile_f32<0>(s2, H, f2, b2, sf, H, lane);
    __syncthreads();
    layernorm16(sf, g, bb, gf + (size_t)tile * 16 * H, H, 16, lane);
}

// ---------------- fused per-edge stage: edge encoder + gather + 4-layer MLP ----
__global__ void k_edge_mlp(
    const float* eattr,
    const float* ew1, const float* eb1,
    const _Float16* ef2, const float* eb2,
    const _Float16* ef3, const float* eb3,
    const float* eg, const float* ebb,
    const int* src, const int* dst, const int* batch,
    const _Float16* h, const _Float16* gf,
    const _Float16* p1f, const float* p1b,
    const _Float16* p2f, const float* p2b,
    const _Float16* p3f, const float* p3b,
    const float* p4w, const float* p4b,
    float* out, int e)
{
    __shared__ _Float16 comb[16 * 512]; // [hsrc | hdst | gf | ef]
    __shared__ _Float16 t1[16 * 256];
    __shared__ _Float16 t2[16 * 128];
    __shared__ _Float16 t3[16 * 64];
    __shared__ float    sf[16 * 128];

    const int tile = blockIdx.x;
    const int lane = threadIdx.x;
    const int m    = lane & 15;
    const int half = (lane >> 4) * 64;
    int eid  = tile * 16 + m;
    int eidc = eid < e ? eid : (e - 1);

    // edge encoder layer 1 (3 -> 128, relu) via VALU; scratch inside t1
    _Float16* s1 = t1;
    _Float16* s2 = t1 + 16 * H;
    float a0 = eattr[(size_t)eidc * 3 + 0];
    float a1 = eattr[(size_t)eidc * 3 + 1];
    float a2 = eattr[(size_t)eidc * 3 + 2];
    for (int c = 0; c < 64; ++c) {
        int cc = half + c;
        float v = a0 * ew1[0 * H + cc] + a1 * ew1[1 * H + cc] + a2 * ew1[2 * H + cc] + eb1[cc];
        s1[m * H + cc] = (_Float16)fmaxf(v, 0.0f);
    }
    __syncthreads();
    gemm_tile_f16<1>(s1, H, ef2, eb2, s2, H, lane);
    __syncthreads();
    gemm_tile_f32<0>(s2, H, ef3, eb3, sf, H, lane);
    __syncthreads();
    layernorm16(sf, eg, ebb, comb + 384, 512, 16, lane); // ef -> comb cols [384,512)

    // gather h[src], h[dst], gf[batch[src]] into comb cols [0,384)
    {
        int s = src[eidc];
        int d = dst[eidc];
        int g = batch[s];
        const _Float16* hs = h  + (size_t)s * H;
        const _Float16* hd = h  + (size_t)d * H;
        const _Float16* gr = gf + (size_t)g * H;
        for (int j = 0; j < 64; j += 8) {
            *(v8h*)&comb[m * 512 +   0 + half + j] = *(const v8h*)&hs[half + j];
            *(v8h*)&comb[m * 512 + 128 + half + j] = *(const v8h*)&hd[half + j];
            *(v8h*)&comb[m * 512 + 256 + half + j] = *(const v8h*)&gr[half + j];
        }
    }
    __syncthreads();

    gemm_tile_f16<2>(comb, 512, p1f, p1b, t1, 256, lane); // ep1 + tanh (16x16 kchunks x 16 ntiles)
    __syncthreads();
    gemm_tile_f16<2>(t1, 256, p2f, p2b, t2, H, lane);     // ep2 + tanh
    __syncthreads();
    gemm_tile_f16<1>(t2, H, p3f, p3b, t3, 64, lane);      // ep3 + relu
    __syncthreads();

    if (lane < 16 && eid < e) {                            // ep4 + sigmoid
        float acc = p4b[0];
        for (int j = 0; j < 64; ++j) acc += (float)t3[lane * 64 + j] * p4w[j];
        out[eid] = 1.0f / (1.0f + expf(-acc));
    }
}

// ---------------- host ----------------
static inline size_t frag_bytes(int K, int Nd)
{
    size_t kc = (size_t)(K + 31) / 32, nt = (size_t)(Nd + 15) / 16;
    return kc * nt * 32 * 16 * sizeof(_Float16);
}

extern "C" void kernel_launch(void* const* d_in, const int* in_sizes, int n_in,
                              void* d_out, int out_size, void* d_ws, size_t ws_size,
                              hipStream_t stream)
{
    (void)in_sizes; (void)n_in; (void)out_size; (void)ws_size;
    const float* x     = (const float*)d_in[0];
    const int*   ei    = (const int*)d_in[1];
    const float* eatt  = (const float*)d_in[2];
    const int*   batch = (const int*)d_in[3];
    const float *ne1_w=(const float*)d_in[4],  *ne1_b=(const float*)d_in[5];
    const float *ne2_w=(const float*)d_in[6],  *ne2_b=(const float*)d_in[7];
    const float *ne3_w=(const float*)d_in[8],  *ne3_b=(const float*)d_in[9];
    const float *ne_g =(const float*)d_in[10], *ne_bb=(const float*)d_in[11];
    const float *ee1_w=(const float*)d_in[12], *ee1_b=(const float*)d_in[13];
    const float *ee2_w=(const float*)d_in[14], *ee2_b=(const float*)d_in[15];
    const float *ee3_w=(const float*)d_in[16], *ee3_b=(const float*)d_in[17];
    const float *ee_g =(const float*)d_in[18], *ee_bb=(const float*)d_in[19];
    const float *g1_w =(const float*)d_in[20], *g1_b =(const float*)d_in[21];
    const float *g2_w =(const float*)d_in[22], *g2_b =(const float*)d_in[23];
    const float *gp1_w=(const float*)d_in[24], *gp1_b=(const float*)d_in[25];
    const float *gp2_w=(const float*)d_in[26], *gp2_b=(const float*)d_in[27];
    const float *gp_g =(const float*)d_in[28], *gp_bb=(const float*)d_in[29];
    const float *ep1_w=(const float*)d_in[30], *ep1_b=(const float*)d_in[31];
    const float *ep2_w=(const float*)d_in[32], *ep2_b=(const float*)d_in[33];
    const float *ep3_w=(const float*)d_in[34], *ep3_b=(const float*)d_in[35];
    const float *ep4_w=(const float*)d_in[36], *ep4_b=(const float*)d_in[37];
    const int* src = ei;
    const int* dst = ei + N_EDGES;
    float* out = (float*)d_out;

    // workspace carve-up (256B aligned)
    char* ws = (char*)d_ws;
    size_t off = 0;
    auto take = [&](size_t bytes) {
        char* p = ws + off;
        off = (off + bytes + 255) & ~(size_t)255;
        return p;
    };
    _Float16* hA   = (_Float16*)take((size_t)N_NODES * H * sizeof(_Float16));
    _Float16* hB   = (_Float16*)take((size_t)N_NODES * H * sizeof(_Float16));
    float*    xw   = (float*)take((size_t)N_NODES * H * sizeof(float));
    float*    agg  = (float*)take((size_t)N_NODES * H * sizeof(float));
    float*    dinv = (float*)take((size_t)N_NODES * sizeof(float));
    float*    sums = (float*)take((size_t)NGRAPH * H * sizeof(float));
    float*    cntg = (float*)take((size_t)NGRAPH * sizeof(float));
    _Float16* gfb  = (_Float16*)take((size_t)NGRAPH * H * sizeof(_Float16));
    _Float16* f_ne2 = (_Float16*)take(frag_bytes(128, 128));
    _Float16* f_ne3 = (_Float16*)take(frag_bytes(128, 128));
    _Float16* f_ee2 = (_Float16*)take(frag_bytes(128, 128));
    _Float16* f_ee3 = (_Float16*)take(frag_bytes(128, 128));
    _Float16* f_g1  = (_Float16*)take(frag_bytes(128, 128));
    _Float16* f_g2  = (_Float16*)take(frag_bytes(128, 128));
    _Float16* f_gp1 = (_Float16*)take(frag_bytes(128, 128));
    _Float16* f_gp2 = (_Float16*)take(frag_bytes(128, 128));
    _Float16* f_p1  = (_Float16*)take(frag_bytes(512, 256));
    _Float16* f_p2  = (_Float16*)take(frag_bytes(256, 128));
    _Float16* f_p3  = (_Float16*)take(frag_bytes(128, 64));

    auto prep = [&](const float* W, _Float16* frag, int K, int Nd) {
        int kc = (K + 31) / 32, nt = (Nd + 15) / 16;
        int total = kc * nt * 32;
        k_prep_bfrag<<<(total + 255) / 256, 256, 0, stream>>>(W, frag, K, Nd);
    };
    auto zero = [&](float* p, long cnt) {
        k_zero<<<(int)((cnt + 255) / 256), 256, 0, stream>>>(p, cnt);
    };

    // 1) weight fragments
    prep(ne2_w, f_ne2, 128, 128); prep(ne3_w, f_ne3, 128, 128);
    prep(ee2_w, f_ee2, 128, 128); prep(ee3_w, f_ee3, 128, 128);
    prep(g1_w,  f_g1,  128, 128); prep(g2_w,  f_g2,  128, 128);
    prep(gp1_w, f_gp1, 128, 128); prep(gp2_w, f_gp2, 128, 128);
    prep(ep1_w, f_p1,  512, 256); prep(ep2_w, f_p2,  256, 128);
    prep(ep3_w, f_p3,  128, 64);

    // 2) degrees -> dinv
    zero(dinv, N_NODES);
    k_deg<<<(N_EDGES + 255) / 256, 256, 0, stream>>>(dst, dinv, N_EDGES);
    k_dinv<<<(N_NODES + 255) / 256, 256, 0, stream>>>(dinv, N_NODES);

    // 3) node encoder -> hA (f16)
    const int ntile = (N_NODES + 15) / 16;
    k_node_enc<<<ntile, 32, 0, stream>>>(x, ne1_w, ne1_b, f_ne2, ne2_b, f_ne3, ne3_b,
                                         ne_g, ne_bb, hA, N_NODES);

    // 4) GCN layer 1: xw = hA@g1 ; scatter ; finish(relu) -> hB
    k_gemm_h<<<ntile, 32, 0, stream>>>(hA, f_g1, xw);
    zero(agg, (long)N_NODES * H);
    k_scatter<<<(int)(((long)N_EDGES * 32 + 255) / 256), 256, 0, stream>>>(src, dst, xw, dinv, agg, N_EDGES);
    k_gcn_fin<<<(int)(((long)N_NODES * H + 255) / 256), 256, 0, stream>>>(agg, xw, dinv, g1_b, hB, N_NODES, 1);

    // 5) GCN layer 2 (no relu) -> hA (final node features, f16)
    k_gemm_h<<<ntile, 32, 0, stream>>>(hB, f_g2, xw);
    zero(agg, (long)N_NODES * H);
    k_scatter<<<(int)(((long)N_EDGES * 32 + 255) / 256), 256, 0, stream>>>(src, dst, xw, dinv, agg, N_EDGES);
    k_gcn_fin<<<(int)(((long)N_NODES * H + 255) / 256), 256, 0, stream>>>(agg, xw, dinv, g2_b, hA, N_NODES, 0);

    // 6) global mean pool + graph MLP -> gfb (f16 [G,128])
    zero(sums, (long)NGRAPH * H);
    zero(cntg, NGRAPH);
    k_pool<<<(int)(((long)N_NODES * 32 + 255) / 256), 256, 0, stream>>>(hA, batch, sums, cntg, N_NODES);
    k_graph<<<NGRAPH / 16, 32, 0, stream>>>(sums, cntg, f_gp1, gp1_b, f_gp2, gp2_b, gp_g, gp_bb, gfb);

    // 7) fused edge encoder + edge-score MLP (dominant stage, WMMA-dense)
    const int etile = (N_EDGES + 15) / 16;
    k_edge_mlp<<<etile, 32, 0, stream>>>(eatt,
                                         ee1_w, ee1_b, f_ee2, ee2_b, f_ee3, ee3_b, ee_g, ee_bb,
                                         src, dst, batch, hA, gfb,
                                         f_p1, ep1_b, f_p2, ep2_b, f_p3, ep3_b, ep4_w, ep4_b,
                                         out, N_EDGES);
}
